// QM9_20486994002063
// MI455X (gfx1250) — matrix operand order repaired and checked by reading, not compiled, not run on hardware
//
#include <hip/hip_runtime.h>
#include <stdint.h>

// ---------------------------------------------------------------------------
// EGAT GNN (QM9-style) for MI455X / gfx1250, wave32, WMMA bf16 matrix path.
//   * V_WMMA_F32_16X16X32_BF16 GEMMs; weights pre-packed to B lane layout
//   * native f32->bf16 converts; b128 streaming A loads; fused bias+residual
//   * fused leaky-relu + attention-score kernel (wave shuffle reduction)
//   * segment softmax via order-preserving uint atomicMax + float atomics
// ---------------------------------------------------------------------------

typedef __attribute__((ext_vector_type(16))) __bf16          v16bf;
typedef __attribute__((ext_vector_type(8)))  float           v8f;

// ---- pack W[K x N] f32 row-major -> bf16 in WMMA B-operand order:
// P[((kb*NT + t)*32 + lane)*16 + j] = bf16( W[(kb*32 + (lane>=16?8:0) + (j<8?j:j+8))*N + t*16 + (lane&15)] )
__global__ void packW_kernel(const float* __restrict__ W, __bf16* __restrict__ P,
                             int K, int N)
{
    int idx = blockIdx.x * blockDim.x + threadIdx.x;
    int total = K * N;
    if (idx >= total) return;
    int NT = N >> 4;
    int j    = idx & 15;
    int lane = (idx >> 4) & 31;
    int tt   = (idx >> 9) % NT;
    int kb   = (idx >> 9) / NT;
    int k = kb * 32 + ((lane >= 16) ? 8 : 0) + ((j < 8) ? j : j + 8);
    int n = tt * 16 + (lane & 15);
    P[idx] = (__bf16)W[(long long)k * N + n];
}

// ---------------- WMMA GEMM:  C[M x NDIM] = A[M x KDIM] @ W + bias (+res)
// 8 waves / block; each wave owns MT consecutive 16-row tiles and reuses the
// B-chunk registers across them. M must be a multiple of 16 (50000, 400000).
template <int KDIM, int NDIM, int MT>
__global__ __launch_bounds__(256) void wmma_gemm_kernel(
    const float* __restrict__ A, const __bf16* __restrict__ Wp,
    const float* __restrict__ bias, const float* __restrict__ res,
    float* __restrict__ C, int M)
{
    constexpr int NT = NDIM / 16;
    constexpr int KB = KDIM / 32;
    const int lane = threadIdx.x & 31;
    const int wave = threadIdx.x >> 5;
    const int half = lane >> 4;       // 0: lanes 0-15, 1: lanes 16-31
    const int l15  = lane & 15;
    const int Mtiles = M >> 4;
    const int t0 = (blockIdx.x * 8 + wave) * MT;   // first row-tile of this wave
    if (t0 >= Mtiles) return;                      // whole-wave (uniform) exit

    v8f acc[MT][NT];
    #pragma unroll
    for (int m = 0; m < MT; ++m)
        #pragma unroll
        for (int t = 0; t < NT; ++t)
            #pragma unroll
            for (int r = 0; r < 8; ++r) acc[m][t][r] = 0.0f;

    #pragma unroll
    for (int kb = 0; kb < KB; ++kb) {
        // ---- B chunk: NT tiles, pre-packed bf16, one 32B vector per lane
        v16bf bv[NT];
        #pragma unroll
        for (int t = 0; t < NT; ++t)
            bv[t] = *(const v16bf*)(Wp + ((size_t)(kb * NT + t) * 32 + lane) * 16);

        #pragma unroll
        for (int m = 0; m < MT; ++m) {
            int ti = t0 + m;
            if (ti >= Mtiles) break;               // uniform per wave
            // ---- A tile 16x32 bf16 (ISA 7.12.2): lane covers row ti*16+l15,
            // K runs [kb*32+half*8 .. +7] and [kb*32+half*8+16 .. +23]
            const float* arow = A + ((long long)ti * 16 + l15) * KDIM + kb * 32 + half * 8;
            if (kb + 1 < KB) __builtin_prefetch(arow + 32, 0, 0);
            float4 f0 = *(const float4*)(arow);
            float4 f1 = *(const float4*)(arow + 4);
            float4 f2 = *(const float4*)(arow + 16);
            float4 f3 = *(const float4*)(arow + 20);
            v16bf av;
            av[0] = (__bf16)f0.x; av[1] = (__bf16)f0.y; av[2]  = (__bf16)f0.z; av[3]  = (__bf16)f0.w;
            av[4] = (__bf16)f1.x; av[5] = (__bf16)f1.y; av[6]  = (__bf16)f1.z; av[7]  = (__bf16)f1.w;
            av[8] = (__bf16)f2.x; av[9] = (__bf16)f2.y; av[10] = (__bf16)f2.z; av[11] = (__bf16)f2.w;
            av[12]= (__bf16)f3.x; av[13]= (__bf16)f3.y; av[14] = (__bf16)f3.z; av[15] = (__bf16)f3.w;

            #pragma unroll
            for (int t = 0; t < NT; ++t)
                acc[m][t] = __builtin_amdgcn_wmma_f32_16x16x32_bf16(
                    false, av, false, bv[t], (short)0, acc[m][t], false, false);
        }
    }

    // ---- store: VGPR r -> row = r + 8*half, col = t*16 + l15;  C = acc+bias+res
    #pragma unroll
    for (int m = 0; m < MT; ++m) {
        int ti = t0 + m;
        if (ti >= Mtiles) break;
        #pragma unroll
        for (int t = 0; t < NT; ++t) {
            float badd = bias ? bias[t * 16 + l15] : 0.0f;
            #pragma unroll
            for (int r = 0; r < 8; ++r) {
                long long row = (long long)ti * 16 + r + (half ? 8 : 0);
                long long pos = row * NDIM + t * 16 + l15;
                float v = acc[m][t][r] + badd;
                if (res) v += res[pos];
                C[pos] = v;
            }
        }
    }
}

// ---------------- small helpers ----------------
__device__ __forceinline__ void mat3mul(const float* a, const float* b, float* c) {
    for (int i = 0; i < 3; ++i)
        for (int j = 0; j < 3; ++j)
            c[i * 3 + j] = a[i * 3 + 0] * b[0 * 3 + j] + a[i * 3 + 1] * b[1 * 3 + j] +
                           a[i * 3 + 2] * b[2 * 3 + j];
}

// R[0..8] = rotmat(angles_rot);  R[9..17] = -rotmat(angles_ref)
__global__ void rotmats_kernel(const float* __restrict__ ar, const float* __restrict__ af,
                               float* __restrict__ R)
{
    if (threadIdx.x != 0 || blockIdx.x != 0) return;
    for (int w = 0; w < 2; ++w) {
        const float* a = w ? af : ar;
        float t = a[0], p = a[1], q = a[2];
        float ct = cosf(t), st = sinf(t);
        float cp = cosf(p), sp = sinf(p);
        float cq = cosf(q), sq = sinf(q);
        float Rt[9] = {ct, -st, 0.f, st, ct, 0.f, 0.f, 0.f, 1.f};
        float Rp[9] = {cp, 0.f, -sp, 0.f, 1.f, 0.f, sp, 0.f, cp};
        float Rq[9] = {1.f, 0.f, 0.f, 0.f, cq, -sq, 0.f, sq, cq};
        float M1[9], M2[9];
        mat3mul(Rt, Rp, M1);
        mat3mul(M1, Rq, M2);
        float s = w ? -1.f : 1.f;
        for (int i = 0; i < 9; ++i) R[w * 9 + i] = s * M2[i];
    }
}

// out[e,j] = sum_i (efeats[e,:] @ R[which])[i] * W1[i,j] + b1[j]   (K=3, no WMMA)
__global__ void edge_ee1_kernel(const float* __restrict__ ef, const float* __restrict__ R,
                                int which, const float* __restrict__ W1,
                                const float* __restrict__ b1, float* __restrict__ out, int E)
{
    long long idx = (long long)blockIdx.x * blockDim.x + threadIdx.x;
    if (idx >= (long long)E * 64) return;
    long long e = idx >> 6;
    int j = idx & 63;
    const float* Rw = R + which * 9;
    float x0 = ef[e * 3], x1 = ef[e * 3 + 1], x2 = ef[e * 3 + 2];
    float s = b1[j];
    #pragma unroll
    for (int i = 0; i < 3; ++i) {
        float r = x0 * Rw[0 * 3 + i] + x1 * Rw[1 * 3 + i] + x2 * Rw[2 * 3 + i];
        s += r * W1[i * 64 + j];
    }
    out[idx] = s;
}

// nh[i,:] = emb_table[ntype[i], :]
__global__ void embed_kernel(const int* __restrict__ ntype, const float* __restrict__ table,
                             float* __restrict__ out, int N)
{
    long long idx = (long long)blockIdx.x * blockDim.x + threadIdx.x;
    if (idx >= (long long)N * 64) return;
    long long i = idx >> 6;
    int c = idx & 63;
    out[idx] = table[(long long)ntype[i] * 64 + c];
}

// hierarchical per-column sum/sumsq -> stats[0..nc-1]=sum, stats[nc..2nc-1]=sumsq
__global__ void colstats_kernel(const float* __restrict__ x, float* __restrict__ stats,
                                int M, int nc)
{
    __shared__ float ss[256], sq[256];
    int rpt = 256 / nc;                 // nc is 64 or 128
    int c = threadIdx.x % nc;
    int rs = threadIdx.x / nc;
    const int RPB = 1024;
    long long r0 = (long long)blockIdx.x * RPB;
    float s = 0.f, s2 = 0.f;
    for (int i = rs; i < RPB; i += rpt) {
        long long r = r0 + i;
        if (r < M) {
            float v = x[r * nc + c];
            s += v;
            s2 += v * v;
        }
    }
    ss[threadIdx.x] = s;
    sq[threadIdx.x] = s2;
    __syncthreads();
    if (rs == 0) {
        for (int i = 1; i < rpt; ++i) { s += ss[i * nc + c]; s2 += sq[i * nc + c]; }
        atomicAdd(&stats[c], s);
        atomicAdd(&stats[nc + c], s2);
    }
}

// y = g*(x-mu)*rsqrt(var+1e-5)+b  [+relu | +prelu]  [optionally accumulate into y]
__global__ void bn_apply_kernel(const float* __restrict__ x, float* __restrict__ y,
                                const float* __restrict__ stats, const float* __restrict__ g,
                                const float* __restrict__ b, int M, int nc, int flags,
                                const float* __restrict__ prelu, int accum)
{
    long long idx = (long long)blockIdx.x * blockDim.x + threadIdx.x;
    long long tot = (long long)M * nc;
    if (idx >= tot) return;
    int c = (int)(idx % nc);
    float invM = 1.0f / (float)M;
    float mu = stats[c] * invM;
    float var = stats[nc + c] * invM - mu * mu;
    float v = g[c] * (x[idx] - mu) * rsqrtf(var + 1e-5f) + b[c];
    if (flags & 1) v = fmaxf(v, 0.f);
    if (flags & 2) v = (v >= 0.f) ? v : prelu[0] * v;
    y[idx] = accum ? (y[idx] + v) : v;
}

// Fused: f_out = leaky_relu(efW + Wni_x[src] + Wnj_x[dst]) (in place on efW)
//        es[e,h] = sum_d f_out[e,h,d] * attn[h,d]    (wave shuffle reduction)
// one wave per (edge, head); lane covers 2 contiguous features.
__global__ __launch_bounds__(256) void edge_fout_att_kernel(
    float* __restrict__ efw, const float* __restrict__ ni, const float* __restrict__ nj,
    const float* __restrict__ attn, const int* __restrict__ src, const int* __restrict__ dst,
    float* __restrict__ es, int E, int H)
{
    long long gw = (long long)blockIdx.x * 8 + (threadIdx.x >> 5);   // (e,h) index
    int lane = threadIdx.x & 31;
    if (gw >= (long long)E * H) return;
    long long e = gw / H;
    int h = (int)(gw % H);
    int HD = H * 64;
    int j0 = h * 64 + lane * 2;
    long long base = e * HD + j0;

    float2 ev  = *(const float2*)(efw + base);
    float2 niv = *(const float2*)(ni + (long long)src[e] * HD + j0);
    float2 njv = *(const float2*)(nj + (long long)dst[e] * HD + j0);
    float v0 = ev.x + niv.x + njv.x;
    float v1 = ev.y + niv.y + njv.y;
    v0 = (v0 > 0.f) ? v0 : 0.01f * v0;
    v1 = (v1 > 0.f) ? v1 : 0.01f * v1;
    float2 o2; o2.x = v0; o2.y = v1;
    *(float2*)(efw + base) = o2;

    float s = v0 * attn[h * 64 + lane * 2] + v1 * attn[h * 64 + lane * 2 + 1];
    #pragma unroll
    for (int o = 16; o > 0; o >>= 1) s += __shfl_xor(s, o, 32);
    if (lane == 0) es[gw] = s;
}

// order-preserving float<->uint map for atomic max
__device__ __forceinline__ unsigned fkey(float x) {
    unsigned u = __float_as_uint(x);
    return (u & 0x80000000u) ? ~u : (u | 0x80000000u);
}
__device__ __forceinline__ float funkey(unsigned k) {
    unsigned u = (k & 0x80000000u) ? (k & 0x7FFFFFFFu) : ~k;
    return __uint_as_float(u);
}

__global__ void segmax_kernel(const float* __restrict__ es, const int* __restrict__ dst,
                              unsigned* __restrict__ nmax, int E, int H)
{
    long long idx = (long long)blockIdx.x * blockDim.x + threadIdx.x;
    if (idx >= (long long)E * H) return;
    long long e = idx / H;
    int h = (int)(idx % H);
    atomicMax(&nmax[(long long)dst[e] * H + h], fkey(es[idx]));
}

__global__ void expsum_kernel(float* __restrict__ es, const int* __restrict__ dst,
                              const unsigned* __restrict__ nmax, float* __restrict__ nsum,
                              int E, int H)
{
    long long idx = (long long)blockIdx.x * blockDim.x + threadIdx.x;
    if (idx >= (long long)E * H) return;
    long long e = idx / H;
    int h = (int)(idx % H);
    float v = __expf(es[idx] - funkey(nmax[(long long)dst[e] * H + h]));
    es[idx] = v;
    atomicAdd(&nsum[(long long)dst[e] * H + h], v);
}

__global__ void anorm_kernel(float* __restrict__ es, const int* __restrict__ dst,
                             const float* __restrict__ nsum, int E, int H)
{
    long long idx = (long long)blockIdx.x * blockDim.x + threadIdx.x;
    if (idx >= (long long)E * H) return;
    long long e = idx / H;
    int h = (int)(idx % H);
    es[idx] = es[idx] / nsum[(long long)dst[e] * H + h];
}

// out[dst[e], j] += hsrc[src[e], j] * a[e, j/64]
__global__ void agg_kernel(const float* __restrict__ hsrc, const float* __restrict__ a,
                           const int* __restrict__ src, const int* __restrict__ dst,
                           float* __restrict__ out, int E, int H, int HD)
{
    long long idx = (long long)blockIdx.x * blockDim.x + threadIdx.x;
    if (idx >= (long long)E * HD) return;
    long long e = idx / HD;
    int j = (int)(idx % HD);
    float v = hsrc[(long long)src[e] * HD + j] * a[e * H + (j >> 6)];
    atomicAdd(&out[(long long)dst[e] * HD + j], v);
}

// per-node scalar decode + segment-sum over graph_ids
__global__ void decode_out_kernel(const float* __restrict__ h, const float* __restrict__ Wout,
                                  const float* __restrict__ bout, const int* __restrict__ gid,
                                  float* __restrict__ outseg, int N)
{
    int i = blockIdx.x * blockDim.x + threadIdx.x;
    if (i >= N) return;
    float s = bout[0];
    const float* hr = h + (long long)i * 64;
    #pragma unroll
    for (int d = 0; d < 64; ++d) s += hr[d] * Wout[d];
    atomicAdd(&outseg[gid[i]], s);
}

// ---------------------------------------------------------------------------
static inline unsigned cdivl(long long a, long long b) { return (unsigned)((a + b - 1) / b); }

extern "C" void kernel_launch(void* const* d_in, const int* in_sizes, int n_in,
                              void* d_out, int out_size, void* d_ws, size_t ws_size,
                              hipStream_t stream)
{
    const int Nn = in_sizes[0];   // 50000
    const int E  = in_sizes[1];   // 400000
    const int G  = 512;

    const int*   ntype  = (const int*)d_in[0];
    const int*   src    = (const int*)d_in[1];
    const int*   dst    = (const int*)d_in[2];
    const int*   gid    = (const int*)d_in[3];
    const float* a_rot  = (const float*)d_in[4];
    const float* a_ref  = (const float*)d_in[5];
    const float* efeats = (const float*)d_in[6];
    const float* emb    = (const float*)d_in[7];
    const float* enc_W  = (const float*)d_in[8];
    const float* enc_b  = (const float*)d_in[9];
    const float* ee_W1  = (const float*)d_in[10];
    const float* ee_b1  = (const float*)d_in[11];
    const float* ee_W2  = (const float*)d_in[12];
    const float* ee_b2  = (const float*)d_in[13];
    const float* enc_bn1_g = (const float*)d_in[14];
    const float* enc_bn1_b = (const float*)d_in[15];
    const float* enc_bn2_g = (const float*)d_in[16];
    const float* enc_bn2_b = (const float*)d_in[17];
    const float* ee_bn1_g  = (const float*)d_in[18];
    const float* ee_bn1_b  = (const float*)d_in[19];
    const float* ee_bn2_g  = (const float*)d_in[20];
    const float* ee_bn2_b  = (const float*)d_in[21];
    // l0 params at 22..35, l1 params at 36..49
    const float* dec_W    = (const float*)d_in[50];
    const float* dec_b    = (const float*)d_in[51];
    const float* dec_bn_g = (const float*)d_in[52];
    const float* dec_bn_b = (const float*)d_in[53];
    const float* dec_prel = (const float*)d_in[54];
    const float* dec_Wout = (const float*)d_in[55];
    const float* dec_bout = (const float*)d_in[56];

    // ---------------- workspace layout (floats; all region sizes 32B-aligned)
    float* ws = (float*)d_ws;
    size_t off = 0;
    auto walloc = [&](size_t n) { float* p = ws + off; off += n; return p; };
    float* R     = walloc(32);
    float* stats = walloc(256);
    float* PWf   = walloc(8192);             // packed bf16 weights (<=16384 elems)
    float* nmax  = walloc((size_t)Nn * 2);
    float* nsum  = walloc((size_t)Nn * 2);
    float* NA  = walloc((size_t)Nn * 64);    // node hidden nh
    float* NT1 = walloc((size_t)Nn * 128);   // Wni out
    float* NT2 = walloc((size_t)Nn * 128);   // Wnj out
    float* NT3 = walloc((size_t)Nn * 128);   // Wnode out (hsrc source)
    float* NT4 = walloc((size_t)Nn * 128);   // aggregation / decoder temp
    float* NT5 = walloc((size_t)Nn * 64);    // gemm out temp
    float* EA  = walloc((size_t)E * 64);     // edge hidden eh
    float* ET1 = walloc((size_t)E * 128);    // Wfij out -> f_out -> bn
    float* ET2 = walloc((size_t)E * 64);     // edge encoder temp
    float* ES  = walloc((size_t)E * 2);      // attention scores / weights
    __bf16* PW = (__bf16*)PWf;
    (void)ws_size; (void)n_in; (void)out_size;

    const int TB = 256;

    auto gemm = [&](const float* A, const float* W, const float* bias, const float* res,
                    float* C, int M, int K, int Ncols) {
        packW_kernel<<<cdivl((long long)K * Ncols, TB), TB, 0, stream>>>(W, PW, K, Ncols);
        int Mtiles = M / 16;
        if (K == 64 && Ncols == 64) {
            wmma_gemm_kernel<64, 64, 4><<<cdivl(Mtiles, 4 * 8), 256, 0, stream>>>(A, PW, bias, res, C, M);
        } else if (K == 64 && Ncols == 128) {
            wmma_gemm_kernel<64, 128, 2><<<cdivl(Mtiles, 2 * 8), 256, 0, stream>>>(A, PW, bias, res, C, M);
        } else { // 128 x 64
            wmma_gemm_kernel<128, 64, 4><<<cdivl(Mtiles, 4 * 8), 256, 0, stream>>>(A, PW, bias, res, C, M);
        }
    };

    auto bnorm = [&](const float* x, float* y, const float* g, const float* b,
                     long long M, int nc, int flags, const float* prelu, int accum) {
        hipMemsetAsync(stats, 0, 2 * nc * sizeof(float), stream);
        colstats_kernel<<<cdivl(M, 1024), TB, 0, stream>>>(x, stats, (int)M, nc);
        bn_apply_kernel<<<cdivl(M * nc, TB), TB, 0, stream>>>(x, y, stats, g, b, (int)M, nc,
                                                              flags, prelu, accum);
    };

    // ---------------- rotation matrices ----------------
    rotmats_kernel<<<1, 32, 0, stream>>>(a_rot, a_ref, R);

    // ---------------- node encoder ----------------
    embed_kernel<<<cdivl((long long)Nn * 64, TB), TB, 0, stream>>>(ntype, emb, NA, Nn);
    bnorm(NA, NA, enc_bn1_g, enc_bn1_b, Nn, 64, /*relu*/1, nullptr, 0);
    gemm(NA, enc_W, enc_b, nullptr, NT5, Nn, 64, 64);
    bnorm(NT5, NA, enc_bn2_g, enc_bn2_b, Nn, 64, 1, nullptr, 0);

    // ---------------- edge encoder (rot branch + reflected branch, summed) ----
    for (int br = 0; br < 2; ++br) {
        edge_ee1_kernel<<<cdivl((long long)E * 64, TB), TB, 0, stream>>>(
            efeats, R, br, ee_W1, ee_b1, ET2, E);
        bnorm(ET2, ET2, ee_bn1_g, ee_bn1_b, E, 64, 1, nullptr, 0);
        gemm(ET2, ee_W2, ee_b2, nullptr, ET1, E, 64, 64);
        bnorm(ET1, EA, ee_bn2_g, ee_bn2_b, E, 64, 1, nullptr, /*accum into eh*/ br);
    }

    // ---------------- EGAT layer ----------------
    auto layer = [&](int base, int H) {
        const float* Wnode = (const float*)d_in[base + 0];
        const float* bnode = (const float*)d_in[base + 1];
        const float* Wni   = (const float*)d_in[base + 2];
        const float* Wnj   = (const float*)d_in[base + 3];
        const float* Wfij  = (const float*)d_in[base + 4];
        const float* attn  = (const float*)d_in[base + 5];
        const float* bnn_g = (const float*)d_in[base + 6];
        const float* bnn_b = (const float*)d_in[base + 7];
        const float* bne_g = (const float*)d_in[base + 8];
        const float* bne_b = (const float*)d_in[base + 9];
        const float* Wpn   = (const float*)d_in[base + 10];
        const float* bpn   = (const float*)d_in[base + 11];
        const float* Wpe   = (const float*)d_in[base + 12];
        const float* bpe   = (const float*)d_in[base + 13];
        const int HD = H * 64;

        gemm(NA, Wni, nullptr, nullptr, NT1, Nn, 64, HD);
        gemm(NA, Wnj, nullptr, nullptr, NT2, Nn, 64, HD);
        gemm(NA, Wnode, bnode, nullptr, NT3, Nn, 64, HD);
        gemm(EA, Wfij, nullptr, nullptr, ET1, E, 64, HD);

        long long thd = (long long)E * HD, th = (long long)E * H;
        // fused leaky-relu + attention score (one wave per (e,h))
        edge_fout_att_kernel<<<cdivl(th, 8), 256, 0, stream>>>(
            ET1, NT1, NT2, attn, src, dst, ES, E, H);

        hipMemsetAsync(nmax, 0, (size_t)Nn * H * sizeof(float), stream); // key 0 == -inf floor
        segmax_kernel<<<cdivl(th, TB), TB, 0, stream>>>(ES, dst, (unsigned*)nmax, E, H);
        hipMemsetAsync(nsum, 0, (size_t)Nn * H * sizeof(float), stream);
        expsum_kernel<<<cdivl(th, TB), TB, 0, stream>>>(ES, dst, (const unsigned*)nmax, nsum, E, H);
        anorm_kernel<<<cdivl(th, TB), TB, 0, stream>>>(ES, dst, nsum, E, H);

        hipMemsetAsync(NT4, 0, (size_t)Nn * HD * sizeof(float), stream);
        agg_kernel<<<cdivl(thd, TB), TB, 0, stream>>>(NT3, ES, src, dst, NT4, E, H, HD);

        // node path: bn -> Wpn (+bpn, +nh residual, in place into NA)
        bnorm(NT4, NT4, bnn_g, bnn_b, Nn, HD, 0, nullptr, 0);
        gemm(NT4, Wpn, bpn, NA, NA, Nn, HD, 64);

        // edge path: bn(f_out) -> Wpe (+bpe, +eh residual, in place into EA)
        bnorm(ET1, ET1, bne_g, bne_b, E, HD, 0, nullptr, 0);
        gemm(ET1, Wpe, bpe, EA, EA, E, HD, 64);
    };

    float* out_h1 = (float*)d_out;
    float* out_h2 = out_h1 + (size_t)Nn * 64;
    float* out_sg = out_h2 + (size_t)Nn * 64;

    layer(22, 2);
    hipMemcpyAsync(out_h1, NA, (size_t)Nn * 64 * sizeof(float),
                   hipMemcpyDeviceToDevice, stream);
    layer(36, 1);
    hipMemcpyAsync(out_h2, NA, (size_t)Nn * 64 * sizeof(float),
                   hipMemcpyDeviceToDevice, stream);

    // ---------------- decoder ----------------
    const float* cur = NA;
    float* pp[2] = {NT5, NT4};
    for (int i = 0; i < 4; ++i) {
        float* o = pp[i & 1];
        gemm(cur, dec_W + (size_t)i * 64 * 64, dec_b + i * 64, nullptr, o, Nn, 64, 64);
        bnorm(o, o, dec_bn_g + i * 64, dec_bn_b + i * 64, Nn, 64, /*prelu*/2, dec_prel + i, 0);
        cur = o;
    }
    hipMemsetAsync(out_sg, 0, G * sizeof(float), stream);
    decode_out_kernel<<<cdivl(Nn, TB), TB, 0, stream>>>(cur, dec_Wout, dec_bout, gid, out_sg, Nn);
}
